// MoELayer_61907658605190
// MI455X (gfx1250) — matrix operand order repaired and checked by reading, not compiled, not run on hardware
//
#include <hip/hip_runtime.h>
#include <hip/hip_bf16.h>
#include <math.h>

// ---------------------------------------------------------------------------
// MoE layer for MI455X (gfx1250): sparse top-2 routing + bf16 WMMA FFN.
// D_MODEL=1024, D_FF=4096, E=8, TOP_K=2, tokens = 2*2048 = 4096.
// FFN block: 32 tokens, 512 threads (16 waves), ~241KB LDS.
// ---------------------------------------------------------------------------

#define D_MODEL   1024
#define D_FF      4096
#define NEXPERTS  8
#define NTOK      4096
#define CAP       4096            // per-expert token list capacity
#define TILE_M    32              // tokens per FFN block (2 x 16-row WMMA tiles)
#define FF_CHUNK  128             // Dff chunk per iteration
#define KC        64              // K slab for W1/W2 staging

// LDS strides (in bf16 elements). Chosen so stride_in_dwords == 4 (mod 32):
// conflict-friendly b128 fragment loads, 16B-aligned rows, 8B-aligned b64 stores.
#define XS_STRIDE  1032           // 32 rows x 1024 cols (x tile)
#define W1S_STRIDE 72             // 128 cols x 64 k     (transposed)
#define HS_STRIDE  136            // 32 rows x 128 cols  (h tile)
#define W2S_STRIDE 72             // 1024 cols x 64 k    (transposed)

#define XS_BYTES   (TILE_M * XS_STRIDE * 2)        // 66048
#define W1S_BYTES  (FF_CHUNK * W1S_STRIDE * 2)     // 18432
#define HS_BYTES   (TILE_M * HS_STRIDE * 2)        // 8704
#define W2S_BYTES  (D_MODEL * W2S_STRIDE * 2)      // 147456
#define META_OFF   (XS_BYTES + W1S_BYTES + HS_BYTES + W2S_BYTES)  // 240640
#define LDS_BYTES  (META_OFF + 256)                // 240896

typedef __attribute__((ext_vector_type(8)))  __bf16 v8bf;
typedef __attribute__((ext_vector_type(16))) __bf16 v16bf;
typedef __attribute__((ext_vector_type(8)))  float  v8f;

// ---------------------------------------------------------------------------
// Fragment builders (ISA 7.12.2 layouts, 16-bit 16x16x32).
// A (16x32, MxK): lane<16 -> row=lane,   K {0..7} then {16..23}
//                 lane>=16 -> row=lane-16,K {8..15} then {24..31}
// B (32x16, KxN): lane<16 -> col=lane,   K {0..7} then {16..23}
//                 lane>=16 -> col=lane-16,K {8..15} then {24..31}
// ---------------------------------------------------------------------------
__device__ __forceinline__ v16bf ldA(const __bf16* rowbase, int kbase) {
  const __bf16* p = rowbase + kbase;
  v8bf lo = *reinterpret_cast<const v8bf*>(p);
  v8bf hi = *reinterpret_cast<const v8bf*>(p + 16);
  v16bf r;
#pragma unroll
  for (int i = 0; i < 8; ++i) { r[i] = lo[i]; r[i + 8] = hi[i]; }
  return r;
}

__device__ __forceinline__ v16bf ldB(const __bf16* colbase, int kbase) {
  const __bf16* p = colbase + kbase;
  v8bf lo = *reinterpret_cast<const v8bf*>(p);
  v8bf hi = *reinterpret_cast<const v8bf*>(p + 16);
  v16bf r;
#pragma unroll
  for (int i = 0; i < 8; ++i) { r[i] = lo[i]; r[i + 8] = hi[i]; }
  return r;
}

__device__ __forceinline__ v8f zero8() {
  v8f z;
#pragma unroll
  for (int i = 0; i < 8; ++i) z[i] = 0.f;
  return z;
}

// Pack 4 floats -> 4 bf16, one 8-byte LDS store (pairs fuse to ds_store_2addr_b64).
__device__ __forceinline__ void store4_bf16(__bf16* dst, float a, float b,
                                            float c, float d) {
  union { __bf16 h[4]; unsigned long long u; } pk;
  pk.h[0] = (__bf16)a; pk.h[1] = (__bf16)b; pk.h[2] = (__bf16)c; pk.h[3] = (__bf16)d;
  *reinterpret_cast<unsigned long long*>(dst) = pk.u;
}

// Exact-GELU via branchless Abramowitz-Stegun erf (|err| < 1.5e-7 — far below
// the bf16 operand rounding already present in the matmuls). One v_exp_f32,
// one reciprocal, no EXEC divergence.
__device__ __forceinline__ float gelu_f(float v) {
  float z = v * 0.70710678118654752f;
  float a = fabsf(z);
  float t = __frcp_rn(1.f + 0.3275911f * a);
  float poly = t * (0.254829592f + t * (-0.284496736f + t * (1.421413741f +
               t * (-1.453152027f + t * 1.061405429f))));
  float er = 1.f - poly * __expf(-a * a);
  er = copysignf(er, z);
  return 0.5f * v * (1.f + er);
}

// ---------------------------------------------------------------------------
// Kernel 0: zero output accumulator + per-expert counts.
// ---------------------------------------------------------------------------
__global__ void moe_zero_kernel(float* __restrict__ out, int n, int* __restrict__ counts) {
  int i = blockIdx.x * blockDim.x + threadIdx.x;
  if (i < n) out[i] = 0.f;
  if (i < NEXPERTS) counts[i] = 0;
}

// ---------------------------------------------------------------------------
// Kernel 1: gating. One wave32 per token; softmax over 8 experts, top-2,
// renormalize, scatter token into per-expert lists.
// ---------------------------------------------------------------------------
__global__ void moe_gate_kernel(const float* __restrict__ x,
                                const float* __restrict__ Wg,
                                const float* __restrict__ bg,
                                int* __restrict__ counts,
                                int* __restrict__ tok_id,
                                float* __restrict__ tok_w) {
  const int lane = threadIdx.x & 31;
  const int wv   = threadIdx.x >> 5;
  const int token = blockIdx.x * 8 + wv;
  if (token >= NTOK) return;

  float acc[NEXPERTS];
#pragma unroll
  for (int e = 0; e < NEXPERTS; ++e) acc[e] = 0.f;

  const float* xr = x + (size_t)token * D_MODEL;
  for (int d = lane; d < D_MODEL; d += 32) {
    float xv = xr[d];
    const float4* wg4 = reinterpret_cast<const float4*>(Wg + (size_t)d * NEXPERTS);
    float4 a = wg4[0], b = wg4[1];
    acc[0] += xv * a.x; acc[1] += xv * a.y; acc[2] += xv * a.z; acc[3] += xv * a.w;
    acc[4] += xv * b.x; acc[5] += xv * b.y; acc[6] += xv * b.z; acc[7] += xv * b.w;
  }
#pragma unroll
  for (int off = 16; off > 0; off >>= 1) {
#pragma unroll
    for (int e = 0; e < NEXPERTS; ++e) acc[e] += __shfl_xor(acc[e], off, 32);
  }

  if (lane == 0) {
    float l[NEXPERTS];
    float mx = -3.402823e38f;
#pragma unroll
    for (int e = 0; e < NEXPERTS; ++e) { l[e] = acc[e] + bg[e]; mx = fmaxf(mx, l[e]); }
    float p[NEXPERTS];
#pragma unroll
    for (int e = 0; e < NEXPERTS; ++e) p[e] = __expf(l[e] - mx);
    int i1 = 0; float p1 = p[0];
#pragma unroll
    for (int e = 1; e < NEXPERTS; ++e) if (p[e] > p1) { p1 = p[e]; i1 = e; }
    int i2 = (i1 == 0) ? 1 : 0; float p2 = p[i2];
#pragma unroll
    for (int e = 0; e < NEXPERTS; ++e) if (e != i1 && p[e] > p2) { p2 = p[e]; i2 = e; }
    float inv = 1.f / (p1 + p2);
    int pos1 = atomicAdd(&counts[i1], 1);
    tok_id[i1 * CAP + pos1] = token;
    tok_w [i1 * CAP + pos1] = p1 * inv;
    int pos2 = atomicAdd(&counts[i2], 1);
    tok_id[i2 * CAP + pos2] = token;
    tok_w [i2 * CAP + pos2] = p2 * inv;
  }
}

// ---------------------------------------------------------------------------
// Kernel 2: routed expert FFN.  grid = (CAP/32, NEXPERTS), block = 512 (16 waves).
// Per block: 32 tokens of one expert. y[32x1024] = GELU(x@W1+b1)@W2+b2,
// scaled by gate weight, atomically accumulated into out.
// Wave wv owns m-tile (wv>>3) and 16 (phase1) / 128 (phase2) output columns.
// ---------------------------------------------------------------------------
__global__ void __launch_bounds__(512, 1)
moe_ffn_kernel(const float* __restrict__ x,
               const float* __restrict__ W1,
               const float* __restrict__ b1,
               const float* __restrict__ W2,
               const float* __restrict__ b2,
               const int* __restrict__ counts,
               const int* __restrict__ tok_id,
               const float* __restrict__ tok_w,
               float* __restrict__ out) {
  const int e     = blockIdx.y;
  const int tile0 = blockIdx.x * TILE_M;
  const int cnt   = counts[e];
  if (tile0 >= cnt) return;

  extern __shared__ __align__(16) char smem[];
  __bf16* xs  = reinterpret_cast<__bf16*>(smem);
  __bf16* w1s = reinterpret_cast<__bf16*>(smem + XS_BYTES);
  __bf16* hs  = reinterpret_cast<__bf16*>(smem + XS_BYTES + W1S_BYTES);
  __bf16* w2s = reinterpret_cast<__bf16*>(smem + XS_BYTES + W1S_BYTES + HS_BYTES);
  int*    rts = reinterpret_cast<int*>  (smem + META_OFF);
  float*  rws = reinterpret_cast<float*>(smem + META_OFF + 128);

  const int tid  = threadIdx.x;
  const int lane = tid & 31;
  const int wv   = tid >> 5;           // wave 0..15
  const int mt   = wv >> 3;            // m-tile 0/1 owned by this wave
  const int wn   = wv & 7;             // n-group 0..7
  const int lr   = lane & 15;
  const int lh   = lane >> 4;          // half-wave select

  const float* W1e = W1 + (size_t)e * D_MODEL * D_FF;
  const float* W2e = W2 + (size_t)e * D_FF * D_MODEL;

  // Row metadata (token ids + gate weights) for this tile.
  if (tid < TILE_M) {
    int idx = tile0 + tid;
    if (idx < cnt) { rts[tid] = tok_id[e * CAP + idx]; rws[tid] = tok_w[e * CAP + idx]; }
    else           { rts[tid] = 0;                     rws[tid] = 0.f; }
  }
  __syncthreads();

  // Stage x tile -> bf16 LDS (32 x 1024), row-major, packed b64 stores.
#pragma unroll 4
  for (int i = 0; i < 16; ++i) {
    int lin4 = tid + i * 512;                 // 8192 float4 total
    int r = lin4 >> 8, c = (lin4 & 255) << 2;
    const float4 g = *reinterpret_cast<const float4*>(x + (size_t)rts[r] * D_MODEL + c);
    store4_bf16(xs + r * XS_STRIDE + c, g.x, g.y, g.z, g.w);
  }
  __syncthreads();

  v8f yacc[8];
#pragma unroll
  for (int nt = 0; nt < 8; ++nt) yacc[nt] = zero8();

  for (int ff = 0; ff < D_FF; ff += FF_CHUNK) {
    // hint: pull next W1 chunk toward L2 while we work on this one
    if (ff + FF_CHUNK < D_FF)
      __builtin_prefetch(W1e + (size_t)(tid & 255) * D_FF + ff + FF_CHUNK, 0, 1);

    // ---------------- Phase 1: h = GELU(x @ W1[:, ff:ff+128] + b1) ----------
    v8f hacc = zero8();
    for (int kk = 0; kk < D_MODEL; kk += KC) {
      __syncthreads();   // WAR: previous readers of w1s/w2s done
      // stage W1[kk:kk+64, ff:ff+128] transposed -> w1s[col][k], bf16.
      // 4x4 micro-tiles: 512 tiles, 1 per thread.
      {
        int r0 = (tid >> 5) << 2;           // k base (0..60)
        int c0 = (tid & 31) << 2;           // n base (0..124)
        const float* gp = W1e + (size_t)(kk + r0) * D_FF + ff + c0;
        const float4 g0 = *reinterpret_cast<const float4*>(gp);
        const float4 g1 = *reinterpret_cast<const float4*>(gp + D_FF);
        const float4 g2 = *reinterpret_cast<const float4*>(gp + 2 * D_FF);
        const float4 g3 = *reinterpret_cast<const float4*>(gp + 3 * D_FF);
        store4_bf16(w1s + (c0 + 0) * W1S_STRIDE + r0, g0.x, g1.x, g2.x, g3.x);
        store4_bf16(w1s + (c0 + 1) * W1S_STRIDE + r0, g0.y, g1.y, g2.y, g3.y);
        store4_bf16(w1s + (c0 + 2) * W1S_STRIDE + r0, g0.z, g1.z, g2.z, g3.z);
        store4_bf16(w1s + (c0 + 3) * W1S_STRIDE + r0, g0.w, g1.w, g2.w, g3.w);
      }
      __syncthreads();
      const __bf16* arow = xs + (mt * 16 + lr) * XS_STRIDE;
      const __bf16* bcol = w1s + (wn * 16 + lr) * W1S_STRIDE;
#pragma unroll
      for (int k2 = 0; k2 < KC; k2 += 32) {
        v16bf a = ldA(arow, kk + k2 + lh * 8);
        v16bf b = ldB(bcol, k2 + lh * 8);
        hacc = __builtin_amdgcn_wmma_f32_16x16x32_bf16(
            false, a, false, b, (short)0, hacc, false, false);
      }
    }
    // bias + exact GELU (branchless), write h tile (bf16) to LDS
    {
      const int nl = wn * 16 + lr;
      const float bv = b1[(size_t)e * D_FF + ff + nl];
#pragma unroll
      for (int j = 0; j < 8; ++j) {
        float v = gelu_f(hacc[j] + bv);
        hs[(mt * 16 + j + lh * 8) * HS_STRIDE + nl] = (__bf16)v;
      }
    }
    __syncthreads();

    // ---------------- Phase 2: y += h @ W2[ff:ff+128, :] --------------------
    for (int kh = 0; kh < FF_CHUNK; kh += KC) {
      // stage W2[ff+kh : +64, 0:1024] transposed -> w2s[col][k], bf16.
      // 4x4 micro-tiles: 4096 tiles, 8 per thread.
#pragma unroll 2
      for (int s = 0; s < 8; ++s) {
        int tIdx = tid + s * 512;
        int r0 = (tIdx >> 8) << 2;          // k base (0..60)
        int c0 = (tIdx & 255) << 2;         // n base (0..1020)
        const float* gp = W2e + (size_t)(ff + kh + r0) * D_MODEL + c0;
        const float4 g0 = *reinterpret_cast<const float4*>(gp);
        const float4 g1 = *reinterpret_cast<const float4*>(gp + D_MODEL);
        const float4 g2 = *reinterpret_cast<const float4*>(gp + 2 * D_MODEL);
        const float4 g3 = *reinterpret_cast<const float4*>(gp + 3 * D_MODEL);
        store4_bf16(w2s + (c0 + 0) * W2S_STRIDE + r0, g0.x, g1.x, g2.x, g3.x);
        store4_bf16(w2s + (c0 + 1) * W2S_STRIDE + r0, g0.y, g1.y, g2.y, g3.y);
        store4_bf16(w2s + (c0 + 2) * W2S_STRIDE + r0, g0.z, g1.z, g2.z, g3.z);
        store4_bf16(w2s + (c0 + 3) * W2S_STRIDE + r0, g0.w, g1.w, g2.w, g3.w);
      }
      __syncthreads();
      const __bf16* arow = hs + (mt * 16 + lr) * HS_STRIDE;
#pragma unroll
      for (int nt = 0; nt < 8; ++nt) {
        const __bf16* bcol = w2s + (wn * 128 + nt * 16 + lr) * W2S_STRIDE;
#pragma unroll
        for (int k2 = 0; k2 < KC; k2 += 32) {
          v16bf a = ldA(arow, kh + k2 + lh * 8);
          v16bf b = ldB(bcol, k2 + lh * 8);
          yacc[nt] = __builtin_amdgcn_wmma_f32_16x16x32_bf16(
              false, a, false, b, (short)0, yacc[nt], false, false);
        }
      }
      __syncthreads();   // before next slab overwrites w2s (and next chunk's w1s/hs)
    }
  }

  // ---------------- Epilogue: out += w * (y + b2), atomic scatter -----------
#pragma unroll
  for (int nt = 0; nt < 8; ++nt) {
    const int n = wn * 128 + nt * 16 + lr;
    const float b2v = b2[(size_t)e * D_MODEL + n];
#pragma unroll
    for (int j = 0; j < 8; ++j) {
      const int m = mt * 16 + j + lh * 8;
      if (tile0 + m < cnt) {
        float val = rws[m] * (yacc[nt][j] + b2v);
        atomicAdd(&out[(size_t)rts[m] * D_MODEL + n], val);
      }
    }
  }
}

// ---------------------------------------------------------------------------
// Launch
// ---------------------------------------------------------------------------
extern "C" void kernel_launch(void* const* d_in, const int* in_sizes, int n_in,
                              void* d_out, int out_size, void* d_ws, size_t ws_size,
                              hipStream_t stream) {
  const float* x  = (const float*)d_in[0];
  const float* Wg = (const float*)d_in[1];
  const float* bg = (const float*)d_in[2];
  const float* W1 = (const float*)d_in[3];
  const float* b1 = (const float*)d_in[4];
  const float* W2 = (const float*)d_in[5];
  const float* b2 = (const float*)d_in[6];
  float* out = (float*)d_out;

  char* ws = (char*)d_ws;
  int*   counts = (int*)ws;                                    // 8 ints (256B slot)
  int*   tok_id = (int*)(ws + 256);                            // 8*4096 ints
  float* tok_w  = (float*)(ws + 256 + NEXPERTS * CAP * 4);     // 8*4096 floats

  const int nOut = NTOK * D_MODEL;
  moe_zero_kernel<<<(nOut + 255) / 256, 256, 0, stream>>>(out, nOut, counts);
  moe_gate_kernel<<<NTOK / 8, 256, 0, stream>>>(x, Wg, bg, counts, tok_id, tok_w);
  moe_ffn_kernel<<<dim3(CAP / TILE_M, NEXPERTS), 512, LDS_BYTES, stream>>>(
      x, W1, b1, W2, b2, counts, tok_id, tok_w, out);
}
